// SwinTransformerBlock_48971217109572
// MI455X (gfx1250) — matrix operand order repaired and checked by reading, not compiled, not run on hardware
//
#include <hip/hip_runtime.h>
#include <hip/hip_bf16.h>

// ---------------------------------------------------------------------------
// SwinV2 block for MI455X (gfx1250): all GEMMs on v_wmma_f32_16x16x32_f16,
// LDS staging via TDM / async-to-LDS where the toolchain exposes them.
// ---------------------------------------------------------------------------

#define DEV __device__ __forceinline__

typedef _Float16 v16h __attribute__((ext_vector_type(16)));
typedef _Float16 v8h  __attribute__((ext_vector_type(8)));
typedef float    v8f  __attribute__((ext_vector_type(8)));
typedef unsigned int v4u __attribute__((ext_vector_type(4)));
typedef int      v4i __attribute__((ext_vector_type(4)));
typedef int      v8i __attribute__((ext_vector_type(8)));

#if defined(__has_builtin)
#  if __has_builtin(__builtin_amdgcn_global_load_async_to_lds_b128)
#    define HAS_ASYNC_LDS 1
#  endif
#  if __has_builtin(__builtin_amdgcn_s_wait_asynccnt)
#    define HAS_WAIT_ASYNC 1
#  endif
#  if __has_builtin(__builtin_amdgcn_tensor_load_to_lds) && \
      __has_builtin(__builtin_amdgcn_s_wait_tensorcnt)
#    define HAS_TDM 1
#  endif
#endif

#if defined(HAS_ASYNC_LDS)
// Parameter types per hipcc diagnostic: int32x4 (vector_size) pointers,
// addrspace(1) source / addrspace(3) LDS destination, non-const.
typedef int i32x4v __attribute__((vector_size(16)));
typedef __attribute__((address_space(1))) i32x4v* gptr_b128_t;
typedef __attribute__((address_space(3))) i32x4v* lptr_b128_t;
#endif

DEV void wait_async_zero() {
#if defined(HAS_WAIT_ASYNC)
  __builtin_amdgcn_s_wait_asynccnt(0);
#elif defined(HAS_ASYNC_LDS)
  asm volatile("s_wait_asynccnt 0x0" ::: "memory");
#endif
}

constexpr int DIM   = 256;
constexpr int HEADS = 8;
constexpr int WS    = 8;
constexpr int SHIFT = 4;
constexpr int HIMG  = 64;
constexpr int WIMG  = 64;
constexpr int NTOK  = WS * WS;          // 64 tokens / window
constexpr int HD    = DIM / HEADS;      // 32
constexpr int BSZ   = 32;
constexpr int NWIN  = (HIMG / WS) * (WIMG / WS);  // 64
constexpr int WINS  = BSZ * NWIN;                 // 2048
constexpr int ROWS  = BSZ * HIMG * WIMG;          // 131072

// ---- WMMA fragment helpers (CDNA5 ISA 7.12.2 layouts, wave32) --------------
DEV v16h load_frag_a(const _Float16* base, int ld, int m0, int k0) {
  const int lane = threadIdx.x & 31;
  const int hlf  = lane >> 4;
  const int m    = m0 + (lane & 15);
  const _Float16* p = base + (size_t)m * ld + k0 + 8 * hlf;
  v8h lo = *(const v8h*)p;
  v8h hi = *(const v8h*)(p + 16);
  v16h r;
#pragma unroll
  for (int i = 0; i < 8; ++i) { r[i] = lo[i]; r[i + 8] = hi[i]; }
  return r;
}

DEV v16h load_frag_b(const _Float16* base, int ld, int n0, int k0) {
  const int lane = threadIdx.x & 31;
  const int hlf  = lane >> 4;
  const int n    = n0 + (lane & 15);
  const _Float16* p = base + (size_t)n * ld + k0 + 8 * hlf;
  v8h lo = *(const v8h*)p;
  v8h hi = *(const v8h*)(p + 16);
  v16h r;
#pragma unroll
  for (int i = 0; i < 8; ++i) { r[i] = lo[i]; r[i + 8] = hi[i]; }
  return r;
}

DEV v8f wmma16(v16h a, v16h b, v8f c) {
  return __builtin_amdgcn_wmma_f32_16x16x32_f16(false, a, false, b, (short)0, c,
                                                false, false);
}

DEV int region9(int g) { return g < 56 ? 0 : (g < 60 ? 1 : 2); }

// ---- TDM: load a 64x256 f16 tile (contiguous rows in memory) into LDS with
//      +8-half row padding (ld = 264) applied by the TDM pad engine. ---------
#if defined(HAS_TDM)
DEV void tdm_load_tile_64x256(const _Float16* src, _Float16* lds_dst) {
  if (threadIdx.x < 32) {  // one wave issues the DMA (EXEC ignored by TDM)
    const unsigned int lds_off = (unsigned int)(size_t)lds_dst;
    const unsigned long long ga = (unsigned long long)(size_t)src;
    v4u g0;
    g0[0] = 1u;                                       // count=1, user mode
    g0[1] = lds_off;                                  // lds_addr
    g0[2] = (unsigned int)(ga & 0xFFFFFFFFu);         // global_addr[31:0]
    g0[3] = (unsigned int)((ga >> 32) & 0x01FFFFFFu)  // global_addr[56:32]
          | (2u << 30);                               // type = 2 ("image")
    v8i g1;
    g1[0] = (int)((1u << 16)      // data_size = 2 bytes
                | (1u << 20)      // pad_enable
                | (6u << 22)      // pad_interval = 128 DWORDs (one 256-h row)
                | (3u << 25));    // pad_amount   = 4 DWORDs (8 halfs)
    g1[1] = (int)((256u & 0xFFFFu) << 16);   // tensor_dim0[15:0] @ [63:48]
    g1[2] = (int)(64u << 16);                // tensor_dim0 hi=0 | tensor_dim1 lo
    g1[3] = (int)(256u << 16);               // tensor_dim1 hi=0 | tile_dim0=256
    g1[4] = (int)(64u);                      // tile_dim1=64 | tile_dim2=0
    g1[5] = (int)(256u);                     // tensor_dim0_stride[31:0]
    g1[6] = 0;                               // stride0 hi | stride1 lo
    g1[7] = 0;                               // stride1 hi
    v4i z4 = {0, 0, 0, 0};
#if __clang_major__ >= 23
    v8i z8 = {0, 0, 0, 0, 0, 0, 0, 0};
    __builtin_amdgcn_tensor_load_to_lds(g0, g1, z4, z4, z8, 0);
#else
    __builtin_amdgcn_tensor_load_to_lds(g0, g1, z4, z4, 0);
#endif
    __builtin_amdgcn_s_wait_tensorcnt(0);
  }
}
#endif

// Fallback / generic staging of a 64x256 f16 tile into ld=264 LDS.
DEV void copy_tile_64x256(const _Float16* src, _Float16* lds_dst, int nthreads) {
  for (int i = threadIdx.x; i < 64 * 256; i += nthreads)
    lds_dst[(i >> 8) * 264 + (i & 255)] = src[i];
}

DEV void stage_tile_64x256(const _Float16* src, _Float16* lds_dst, int nthreads) {
#if defined(HAS_TDM)
  tdm_load_tile_64x256(src, lds_dst);
#else
  copy_tile_64x256(src, lds_dst, nthreads);
#endif
}

// ---------------------------------------------------------------------------
// K0: fp32 -> f16 weight conversion (qkv_w | proj_w | fc1_w | fc2_w)
// ---------------------------------------------------------------------------
__global__ __launch_bounds__(256) void prep_weights(
    const float* __restrict__ qkv_w, const float* __restrict__ proj_w,
    const float* __restrict__ fc1_w, const float* __restrict__ fc2_w,
    _Float16* __restrict__ qkvw_h, _Float16* __restrict__ projw_h,
    _Float16* __restrict__ fc1w_h, _Float16* __restrict__ fc2w_h) {
  const int NQ = 768 * 256, NP = 256 * 256, N1 = 1024 * 256, N2 = 256 * 1024;
  int idx = blockIdx.x * 256 + threadIdx.x;
  int total = NQ + NP + N1 + N2;
  for (; idx < total; idx += gridDim.x * 256) {
    if (idx < NQ)                qkvw_h[idx]           = (_Float16)qkv_w[idx];
    else if (idx < NQ + NP)      projw_h[idx - NQ]     = (_Float16)proj_w[idx - NQ];
    else if (idx < NQ + NP + N1) fc1w_h[idx - NQ - NP] = (_Float16)fc1_w[idx - NQ - NP];
    else                         fc2w_h[idx - NQ - NP - N1] = (_Float16)fc2_w[idx - NQ - NP - N1];
  }
}

// ---------------------------------------------------------------------------
// K1: continuous-position-bias MLP -> 16*sigmoid table (225 x HEADS, fp32)
// ---------------------------------------------------------------------------
__global__ __launch_bounds__(32) void prep_cpb(
    const float* __restrict__ cpb_w1, const float* __restrict__ cpb_b1,
    const float* __restrict__ cpb_w2, float* __restrict__ tab) {
  const int e = blockIdx.x;             // 0..224
  const int i = e / 15, j = e % 15;
  float t0 = (float)(i - 7) / 7.0f * 8.0f;
  float t1 = (float)(j - 7) / 7.0f * 8.0f;
  float c0 = (t0 >= 0.f ? 1.f : -1.f) * log2f(fabsf(t0) + 1.0f) * (1.0f / 3.0f);
  float c1 = (t1 >= 0.f ? 1.f : -1.f) * log2f(fabsf(t1) + 1.0f) * (1.0f / 3.0f);
  float acc[HEADS];
#pragma unroll
  for (int h = 0; h < HEADS; ++h) acc[h] = 0.f;
  for (int u = threadIdx.x; u < 512; u += 32) {
    float hv = fmaxf(c0 * cpb_w1[2 * u] + c1 * cpb_w1[2 * u + 1] + cpb_b1[u], 0.f);
#pragma unroll
    for (int h = 0; h < HEADS; ++h) acc[h] += hv * cpb_w2[h * 512 + u];
  }
#pragma unroll
  for (int h = 0; h < HEADS; ++h) {
#pragma unroll
    for (int m = 16; m >= 1; m >>= 1) acc[h] += __shfl_xor(acc[h], m);
  }
  if (threadIdx.x == 0) {
#pragma unroll
    for (int h = 0; h < HEADS; ++h)
      tab[e * HEADS + h] = 16.0f / (1.0f + __expf(-acc[h]));
  }
}

// ---------------------------------------------------------------------------
// K2: fused shift+window-partition gather + QKV GEMM (64x256 @ 256x768)
// ---------------------------------------------------------------------------
__global__ __launch_bounds__(256) void qkv_kernel(
    const float* __restrict__ x, const _Float16* __restrict__ wq,
    const float* __restrict__ q_bias, const float* __restrict__ v_bias,
    _Float16* __restrict__ qh, _Float16* __restrict__ kh,
    _Float16* __restrict__ vh) {
  __shared__ __align__(16) _Float16 xs[64][264];
  const int win = blockIdx.x;
  const int b = win >> 6, w_in = win & 63;
  const int wr = w_in >> 3, wc = w_in & 7;

  for (int i = threadIdx.x; i < 64 * 256; i += 256) {
    int row = i >> 8, col = i & 255;
    int r = row >> 3, c = row & 7;
    int sr = (wr * 8 + r + SHIFT) & 63;
    int sc = (wc * 8 + c + SHIFT) & 63;
    xs[row][col] = (_Float16)x[((size_t)b * 4096 + sr * 64 + sc) * 256 + col];
  }
  __syncthreads();

  const int wv = threadIdx.x >> 5, lane = threadIdx.x & 31;
  const int rt = wv & 3, ch = wv >> 2;
  const int hlf = lane >> 4, nl = lane & 15;

  // Hoist the 8 A fragments (whole K) for this wave's 16 rows.
  v16h afr[8];
#pragma unroll
  for (int kk = 0; kk < 8; ++kk)
    afr[kk] = load_frag_a(&xs[0][0], 264, rt * 16, kk * 32);

  for (int ct = 0; ct < 24; ++ct) {
    const int n0 = (ch * 24 + ct) * 16;
    __builtin_prefetch(wq + (size_t)(n0 + 16) * 256, 0, 3);
    v8f acc = {};
#pragma unroll
    for (int kb = 0; kb < 8; kb += 4) {           // preload 4 B frags, then MMA
      v16h b4[4];
#pragma unroll
      for (int j = 0; j < 4; ++j) b4[j] = load_frag_b(wq, 256, n0, (kb + j) * 32);
#pragma unroll
      for (int j = 0; j < 4; ++j) acc = wmma16(afr[kb + j], b4[j], acc);
    }
    // Wave-uniform destination select (n0 is uniform -> scalar branches).
    const int feat = n0 + nl;
    _Float16* dst;
    float bias;
    if (n0 < 256)      { dst = qh; bias = q_bias[feat]; }
    else if (n0 < 512) { dst = kh; bias = 0.f; }
    else               { dst = vh; bias = v_bias[feat - 512]; }
    const int hd = (feat >> 5) & 7, d = feat & 31;
    const size_t basei = ((size_t)(win * HEADS + hd) * NTOK) * HD + d;
#pragma unroll
    for (int r = 0; r < 8; ++r) {
      int m = rt * 16 + r + 8 * hlf;
      dst[basei + (size_t)m * HD] = (_Float16)(acc[r] + bias);
    }
  }
}

// ---------------------------------------------------------------------------
// K3: cosine attention per (window, head)
// ---------------------------------------------------------------------------
__global__ __launch_bounds__(128) void attn_kernel(
    const _Float16* __restrict__ qh, const _Float16* __restrict__ kh,
    const _Float16* __restrict__ vh, const float* __restrict__ logit_scale,
    const float* __restrict__ cpb_tab, _Float16* __restrict__ oh) {
  __shared__ __align__(16) _Float16 qs[64][48];
  __shared__ __align__(16) _Float16 ks[64][48];
  __shared__ __align__(16) _Float16 vT[32][80];   // V transposed: vT[d][tok]
  __shared__ __align__(16) _Float16 ps[64][72];   // softmax(P) as next A
  __shared__ float qn[64], kn[64], tab[225];

  const int win = blockIdx.x >> 3;
  const int head = blockIdx.x & 7;
  const size_t base = (size_t)(win * HEADS + head) * NTOK * HD;

#if defined(HAS_ASYNC_LDS)
  // Async DMA of q/k tiles: 16B chunks, per-lane LDS addresses (ASYNCcnt).
#pragma unroll
  for (int it = 0; it < 2; ++it) {
    const int idx = threadIdx.x + it * 128;       // 256 chunks of 8 halfs
    const int row = idx >> 2, c8 = (idx & 3) * 8; // 4 chunks per 32-half row
    __builtin_amdgcn_global_load_async_to_lds_b128(
        (gptr_b128_t)(void*)(qh + base + (size_t)idx * 8),
        (lptr_b128_t)(void*)(&qs[row][c8]), 0, 0);
    __builtin_amdgcn_global_load_async_to_lds_b128(
        (gptr_b128_t)(void*)(kh + base + (size_t)idx * 8),
        (lptr_b128_t)(void*)(&ks[row][c8]), 0, 0);
  }
  for (int i = threadIdx.x; i < NTOK * HD; i += 128) {
    int row = i >> 5, col = i & 31;
    vT[col][row] = vh[base + i];
  }
  wait_async_zero();
#else
  for (int i = threadIdx.x; i < NTOK * HD; i += 128) {
    int row = i >> 5, col = i & 31;
    qs[row][col] = qh[base + i];
    ks[row][col] = kh[base + i];
    vT[col][row] = vh[base + i];
  }
#endif
  for (int i = threadIdx.x; i < 225; i += 128) tab[i] = cpb_tab[i * HEADS + head];
  __syncthreads();

  {  // per-row L2 norms of q and k
    int t = threadIdx.x;
    if (t < 64) {
      float s = 0.f;
#pragma unroll
      for (int c = 0; c < 32; ++c) { float v = (float)qs[t][c]; s += v * v; }
      qn[t] = fmaxf(sqrtf(s), 1e-12f);
    } else {
      int r = t - 64;
      float s = 0.f;
#pragma unroll
      for (int c = 0; c < 32; ++c) { float v = (float)ks[r][c]; s += v * v; }
      kn[r] = fmaxf(sqrtf(s), 1e-12f);
    }
  }
  __syncthreads();

  const float sc = __expf(fminf(logit_scale[head], 4.6051702f));  // ln(100)
  const int wv = threadIdx.x >> 5, lane = threadIdx.x & 31;
  const int hlf = lane >> 4, nl = lane & 15;
  const int m0 = wv * 16;
  const int w_in = win & 63, wr = w_in >> 3, wc = w_in & 7;

  // ---- S = q k^T ----
  v8f s[4];
  {
    v16h a = load_frag_a(&qs[0][0], 48, m0, 0);
#pragma unroll
    for (int j = 0; j < 4; ++j) {
      v16h bf = load_frag_b(&ks[0][0], 48, j * 16, 0);
      v8f z = {};
      s[j] = wmma16(a, bf, z);
    }
  }

  // ---- scale + CPB bias + shift mask + row softmax ----
#pragma unroll
  for (int r = 0; r < 8; ++r) {
    const int m = m0 + r + 8 * hlf;
    const int mr = m >> 3, mc = m & 7;
    const int reg1 = region9(wr * 8 + mr) * 3 + region9(wc * 8 + mc);
    const float qscale = sc / qn[m];
    float vals[4];
    float rmax = -1e30f;
#pragma unroll
    for (int j = 0; j < 4; ++j) {
      const int n = j * 16 + nl;
      const int nr = n >> 3, nc = n & 7;
      float v = s[j][r] * qscale / kn[n];
      v += tab[(mr - nr + 7) * 15 + (mc - nc + 7)];
      const int reg2 = region9(wr * 8 + nr) * 3 + region9(wc * 8 + nc);
      if (reg1 != reg2) v -= 100.0f;
      vals[j] = v;
      rmax = fmaxf(rmax, v);
    }
#pragma unroll
    for (int msk = 8; msk >= 1; msk >>= 1) rmax = fmaxf(rmax, __shfl_xor(rmax, msk));
    float sum = 0.f;
#pragma unroll
    for (int j = 0; j < 4; ++j) { vals[j] = __expf(vals[j] - rmax); sum += vals[j]; }
#pragma unroll
    for (int msk = 8; msk >= 1; msk >>= 1) sum += __shfl_xor(sum, msk);
    const float inv = 1.0f / sum;
#pragma unroll
    for (int j = 0; j < 4; ++j) ps[m][j * 16 + nl] = (_Float16)(vals[j] * inv);
  }
  __syncthreads();

  // ---- O = P V ----
#pragma unroll
  for (int jt = 0; jt < 2; ++jt) {
    v8f acc = {};
#pragma unroll
    for (int k0 = 0; k0 < 64; k0 += 32) {
      v16h a  = load_frag_a(&ps[0][0], 72, m0, k0);
      v16h bf = load_frag_b(&vT[0][0], 80, jt * 16, k0);
      acc = wmma16(a, bf, acc);
    }
#pragma unroll
    for (int r = 0; r < 8; ++r) {
      int m = m0 + r + 8 * hlf;
      int n = jt * 16 + nl;
      oh[((size_t)win * NTOK + m) * DIM + head * HD + n] = (_Float16)acc[r];
    }
  }
}

// ---------------------------------------------------------------------------
// K4: proj GEMM + window-reverse/unshift + post-norm residual
//     xr = shortcut + LN(xo; norm1)     (fp32 + f16 mirror for the MLP)
// ---------------------------------------------------------------------------
__global__ __launch_bounds__(256) void proj_ln_kernel(
    const _Float16* __restrict__ oh, const _Float16* __restrict__ wp,
    const float* __restrict__ proj_b, const float* __restrict__ x,
    const float* __restrict__ n1g, const float* __restrict__ n1b,
    float* __restrict__ xr, _Float16* __restrict__ xrh) {
  extern __shared__ __align__(16) char smem[];
  _Float16* aS = (_Float16*)smem;                     // 64 x 264
  float* outS = (float*)(smem + 64 * 264 * 2);        // 64 x 256 fp32

  const int win = blockIdx.x;
  stage_tile_64x256(oh + (size_t)win * 64 * 256, aS, 256);
  __syncthreads();

  const int wv = threadIdx.x >> 5, lane = threadIdx.x & 31;
  const int rt = wv & 3, ch = wv >> 2;
  const int hlf = lane >> 4, nl = lane & 15;

  v16h afr[8];
#pragma unroll
  for (int kk = 0; kk < 8; ++kk)
    afr[kk] = load_frag_a(aS, 264, rt * 16, kk * 32);

  for (int ct = 0; ct < 8; ++ct) {
    const int n0 = ch * 128 + ct * 16;
    __builtin_prefetch(wp + (size_t)(n0 + 16) * 256, 0, 3);
    v8f acc = {};
#pragma unroll
    for (int kb = 0; kb < 8; kb += 4) {
      v16h b4[4];
#pragma unroll
      for (int j = 0; j < 4; ++j) b4[j] = load_frag_b(wp, 256, n0, (kb + j) * 32);
#pragma unroll
      for (int j = 0; j < 4; ++j) acc = wmma16(afr[kb + j], b4[j], acc);
    }
    const float pb = proj_b[n0 + nl];
#pragma unroll
    for (int r = 0; r < 8; ++r) {
      int m = rt * 16 + r + 8 * hlf;
      outS[m * 256 + n0 + nl] = acc[r] + pb;
    }
  }
  __syncthreads();

  const int t = threadIdx.x;
  const int row = t >> 2, q4 = t & 3;
  float sm = 0.f, ss = 0.f;
  for (int c = q4 * 64; c < q4 * 64 + 64; ++c) {
    float v = outS[row * 256 + c];
    sm += v; ss += v * v;
  }
  sm += __shfl_xor(sm, 1); sm += __shfl_xor(sm, 2);
  ss += __shfl_xor(ss, 1); ss += __shfl_xor(ss, 2);
  const float mean = sm * (1.0f / 256.0f);
  const float var  = ss * (1.0f / 256.0f) - mean * mean;
  const float rstd = rsqrtf(var + 1e-5f);

  const int b = win >> 6, w_in = win & 63;
  const int wr = w_in >> 3, wc = w_in & 7;
  const int rr = row >> 3, cc = row & 7;
  const int dr = (wr * 8 + rr + SHIFT) & 63;
  const int dc = (wc * 8 + cc + SHIFT) & 63;
  const size_t pbase = ((size_t)b * 4096 + dr * 64 + dc) * 256;
  for (int c = q4 * 64; c < q4 * 64 + 64; ++c) {
    float xo = (outS[row * 256 + c] - mean) * rstd * n1g[c] + n1b[c];
    float v = x[pbase + c] + xo;
    xr[pbase + c] = v;
    xrh[pbase + c] = (_Float16)v;
  }
}

// ---------------------------------------------------------------------------
// K5: fused MLP: hdn = GELU(xr @ fc1^T + b1) kept in LDS (f16, 132 KB),
//     y = hdn @ fc2^T + b2, out = xr + LN(y; norm2)
// ---------------------------------------------------------------------------
__global__ __launch_bounds__(256) void mlp_kernel(
    const float* __restrict__ xr, const _Float16* __restrict__ xrh,
    const _Float16* __restrict__ w1, const float* __restrict__ fc1_b,
    const _Float16* __restrict__ w2, const float* __restrict__ fc2_b,
    const float* __restrict__ n2g, const float* __restrict__ n2b,
    float* __restrict__ out) {
  extern __shared__ __align__(16) char smem[];
  _Float16* aS = (_Float16*)smem;                              // 64 x 264
  _Float16* hS = (_Float16*)(smem + 64 * 264 * 2);             // 64 x 1032
  float* outS = (float*)(smem + 64 * 264 * 2 + 64 * 1032 * 2); // 64 x 256 fp32

  const size_t row0 = (size_t)blockIdx.x * 64;
  stage_tile_64x256(xrh + row0 * 256, aS, 256);
  __syncthreads();

  const int wv = threadIdx.x >> 5, lane = threadIdx.x & 31;
  const int rt = wv & 3, ch = wv >> 2;
  const int hlf = lane >> 4, nl = lane & 15;

  v16h afr[8];
#pragma unroll
  for (int kk = 0; kk < 8; ++kk)
    afr[kk] = load_frag_a(aS, 264, rt * 16, kk * 32);

  // ---- fc1 + exact GELU -> hS ----
  for (int ct = 0; ct < 32; ++ct) {
    const int n0 = ch * 512 + ct * 16;
    __builtin_prefetch(w1 + (size_t)(n0 + 16) * 256, 0, 3);
    v8f acc = {};
#pragma unroll
    for (int kb = 0; kb < 8; kb += 4) {
      v16h b4[4];
#pragma unroll
      for (int j = 0; j < 4; ++j) b4[j] = load_frag_b(w1, 256, n0, (kb + j) * 32);
#pragma unroll
      for (int j = 0; j < 4; ++j) acc = wmma16(afr[kb + j], b4[j], acc);
    }
    const float b1 = fc1_b[n0 + nl];
#pragma unroll
    for (int r = 0; r < 8; ++r) {
      int m = rt * 16 + r + 8 * hlf;
      float v = acc[r] + b1;
      float g = 0.5f * v * (1.0f + erff(v * 0.70710678118f));
      hS[m * 1032 + n0 + nl] = (_Float16)g;
    }
  }
  __syncthreads();

  // ---- fc2 -> outS ----
  for (int ct = 0; ct < 8; ++ct) {
    const int n0 = ch * 128 + ct * 16;
    __builtin_prefetch(w2 + (size_t)(n0 + 16) * 1024, 0, 3);
    v8f acc = {};
    for (int kb = 0; kb < 1024; kb += 128) {
      v16h a4[4], b4[4];
#pragma unroll
      for (int j = 0; j < 4; ++j) {
        a4[j] = load_frag_a(hS, 1032, rt * 16, kb + j * 32);
        b4[j] = load_frag_b(w2, 1024, n0, kb + j * 32);
      }
#pragma unroll
      for (int j = 0; j < 4; ++j) acc = wmma16(a4[j], b4[j], acc);
    }
    const float b2 = fc2_b[n0 + nl];
#pragma unroll
    for (int r = 0; r < 8; ++r) {
      int m = rt * 16 + r + 8 * hlf;
      outS[m * 256 + n0 + nl] = acc[r] + b2;
    }
  }
  __syncthreads();

  const int t = threadIdx.x;
  const int row = t >> 2, q4 = t & 3;
  float sm = 0.f, ss = 0.f;
  for (int c = q4 * 64; c < q4 * 64 + 64; ++c) {
    float v = outS[row * 256 + c];
    sm += v; ss += v * v;
  }
  sm += __shfl_xor(sm, 1); sm += __shfl_xor(sm, 2);
  ss += __shfl_xor(ss, 1); ss += __shfl_xor(ss, 2);
  const float mean = sm * (1.0f / 256.0f);
  const float var  = ss * (1.0f / 256.0f) - mean * mean;
  const float rstd = rsqrtf(var + 1e-5f);
  for (int c = q4 * 64; c < q4 * 64 + 64; ++c) {
    float v = (outS[row * 256 + c] - mean) * rstd * n2g[c] + n2b[c];
    out[(row0 + row) * 256 + c] = xr[(row0 + row) * 256 + c] + v;
  }
}

// ---------------------------------------------------------------------------
// Host launcher
// ---------------------------------------------------------------------------
static inline size_t align256(size_t v) { return (v + 255) & ~(size_t)255; }

extern "C" void kernel_launch(void* const* d_in, const int* in_sizes, int n_in,
                              void* d_out, int out_size, void* d_ws, size_t ws_size,
                              hipStream_t stream) {
  (void)in_sizes; (void)n_in; (void)out_size; (void)ws_size;
  const float* x      = (const float*)d_in[0];
  const float* n1g    = (const float*)d_in[1];
  const float* n1b    = (const float*)d_in[2];
  const float* qkv_w  = (const float*)d_in[3];
  const float* q_bias = (const float*)d_in[4];
  const float* v_bias = (const float*)d_in[5];
  const float* logit  = (const float*)d_in[6];
  const float* cpb_w1 = (const float*)d_in[7];
  const float* cpb_b1 = (const float*)d_in[8];
  const float* cpb_w2 = (const float*)d_in[9];
  const float* proj_w = (const float*)d_in[10];
  const float* proj_b = (const float*)d_in[11];
  const float* n2g    = (const float*)d_in[12];
  const float* n2b    = (const float*)d_in[13];
  const float* fc1_w  = (const float*)d_in[14];
  const float* fc1_b  = (const float*)d_in[15];
  const float* fc2_w  = (const float*)d_in[16];
  const float* fc2_b  = (const float*)d_in[17];
  float* out = (float*)d_out;

  char* ws = (char*)d_ws;
  size_t off = 0;
  _Float16* qkvw_h = (_Float16*)(ws + off); off = align256(off + 768 * 256 * 2);
  _Float16* projw_h= (_Float16*)(ws + off); off = align256(off + 256 * 256 * 2);
  _Float16* fc1w_h = (_Float16*)(ws + off); off = align256(off + 1024 * 256 * 2);
  _Float16* fc2w_h = (_Float16*)(ws + off); off = align256(off + 256 * 1024 * 2);
  float*    cpbtab = (float*)   (ws + off); off = align256(off + 225 * 8 * 4);
  const size_t qkv_elems = (size_t)WINS * HEADS * NTOK * HD;   // 33.5M
  _Float16* qh = (_Float16*)(ws + off); off = align256(off + qkv_elems * 2);
  _Float16* kh = (_Float16*)(ws + off); off = align256(off + qkv_elems * 2);
  _Float16* vh = (_Float16*)(ws + off); off = align256(off + qkv_elems * 2);
  _Float16* oh = (_Float16*)(ws + off); off = align256(off + qkv_elems * 2);
  float*    xr = (float*)   (ws + off); off = align256(off + (size_t)ROWS * DIM * 4);
  _Float16* xrh= (_Float16*)(ws + off); off = align256(off + (size_t)ROWS * DIM * 2);

  prep_weights<<<3072, 256, 0, stream>>>(qkv_w, proj_w, fc1_w, fc2_w,
                                         qkvw_h, projw_h, fc1w_h, fc2w_h);
  prep_cpb<<<225, 32, 0, stream>>>(cpb_w1, cpb_b1, cpb_w2, cpbtab);
  qkv_kernel<<<WINS, 256, 0, stream>>>(x, qkvw_h, q_bias, v_bias, qh, kh, vh);
  attn_kernel<<<WINS * HEADS, 128, 0, stream>>>(qh, kh, vh, logit, cpbtab, oh);
  {
    size_t shm = (size_t)64 * 264 * 2 + (size_t)64 * 256 * 4;   // 99 328 B
    proj_ln_kernel<<<WINS, 256, shm, stream>>>(oh, projw_h, proj_b, x, n1g, n1b,
                                               xr, xrh);
  }
  {
    size_t shm = (size_t)64 * 264 * 2 + (size_t)64 * 1032 * 2 +
                 (size_t)64 * 256 * 4;                          // 231 424 B
    mlp_kernel<<<ROWS / 64, 256, shm, stream>>>(xr, xrh, fc1w_h, fc1_b, fc2w_h,
                                                fc2_b, n2g, n2b, out);
  }
}
